// SelfAttentionLayer_9766755631810
// MI455X (gfx1250) — compile-verified, matrix operand order checked by reading, not run on hardware
//
#include <hip/hip_runtime.h>

typedef __attribute__((ext_vector_type(16))) __bf16 v16bf;
typedef __attribute__((ext_vector_type(8)))  __bf16 v8bf;
typedef __attribute__((ext_vector_type(8)))  float  v8f;

#define B_  2
#define S_  2048
#define D_  1024
#define H_  16
#define HD_ 64

// round-to-nearest-even f32 -> bf16 (pure integer, no dependence on cvt ops)
static __device__ __forceinline__ __bf16 f2bf(float f) {
    unsigned u = __builtin_bit_cast(unsigned, f);
    unsigned r = u + 0x7FFFu + ((u >> 16) & 1u);
    unsigned short h = (unsigned short)(r >> 16);
    return __builtin_bit_cast(__bf16, h);
}

// Load one 16-bf16 WMMA fragment: lane holds 16 values along K for its row,
// K order = {g*8..g*8+7, 16+g*8..16+g*8+7}; caller passes p = row_base + g*8.
static __device__ __forceinline__ v16bf ldfrag(const __bf16* p) {
    v8bf lo = *(const v8bf*)p;
    v8bf hi = *(const v8bf*)(p + 16);
    return __builtin_shufflevector(lo, hi, 0,1,2,3,4,5,6,7,8,9,10,11,12,13,14,15);
}

// ---------------------------------------------------------------------------
// Kernel 1: QKV projection. X(4096x1024) @ W(1024x3072) -> q/k/v bf16 (B,H,S,64)
// ---------------------------------------------------------------------------
__global__ __launch_bounds__(256)
void qkv_gemm_wmma(const float* __restrict__ X, const float* __restrict__ W,
                   const float* __restrict__ qb, const float* __restrict__ vb,
                   __bf16* __restrict__ qws, __bf16* __restrict__ kws,
                   __bf16* __restrict__ vws)
{
    __shared__ __bf16 lA[128][40];   // [m][k], stride 80B (16B aligned)
    __shared__ __bf16 lB[128][40];   // [n][k] (B transposed)

    const int t = threadIdx.x;
    const int w = t >> 5, lane = t & 31, g = lane >> 4, nl = lane & 15;
    const int wm = w >> 2, wn = w & 3;            // 2x4 wave grid
    const int rowBase = blockIdx.y * 128;
    const int colBase = blockIdx.x * 128;

    const v8f vzero = {0.f,0.f,0.f,0.f,0.f,0.f,0.f,0.f};
    v8f acc[4][2];
    #pragma unroll
    for (int i = 0; i < 4; i++)
        #pragma unroll
        for (int j = 0; j < 2; j++) acc[i][j] = vzero;

    const int arow = t >> 3, akoff = (t & 7) * 4;   // A: 32 rows x 32 k per chunk
    const int bk   = t >> 5, bnoff = (t & 31) * 4;  // B: 8 k x 128 n per chunk

    for (int k0 = 0; k0 < D_; k0 += 32) {
        __syncthreads();
        #pragma unroll
        for (int c = 0; c < 4; c++) {
            const int r = arow + c * 32;
            const float4 a4 = *(const float4*)(X + (size_t)(rowBase + r) * D_ + k0 + akoff);
            lA[r][akoff + 0] = f2bf(a4.x);
            lA[r][akoff + 1] = f2bf(a4.y);
            lA[r][akoff + 2] = f2bf(a4.z);
            lA[r][akoff + 3] = f2bf(a4.w);
            const int kk = bk + c * 8;
            const float4 b4 = *(const float4*)(W + (size_t)(k0 + kk) * (3 * D_) + colBase + bnoff);
            lB[bnoff + 0][kk] = f2bf(b4.x);
            lB[bnoff + 1][kk] = f2bf(b4.y);
            lB[bnoff + 2][kk] = f2bf(b4.z);
            lB[bnoff + 3][kk] = f2bf(b4.w);
        }
        __syncthreads();

        v16bf bfr[2];
        #pragma unroll
        for (int nt = 0; nt < 2; nt++)
            bfr[nt] = ldfrag(&lB[wn * 32 + nt * 16 + nl][g * 8]);
        #pragma unroll
        for (int mt = 0; mt < 4; mt++) {
            const v16bf af = ldfrag(&lA[wm * 64 + mt * 16 + nl][g * 8]);
            #pragma unroll
            for (int nt = 0; nt < 2; nt++)
                acc[mt][nt] = __builtin_amdgcn_wmma_f32_16x16x32_bf16(
                    false, af, false, bfr[nt], (short)0, acc[mt][nt], false, false);
        }
    }

    // Epilogue: q/k/v routing is BLOCK-UNIFORM (colBase multiple of 128,
    // q/k/v boundaries at multiples of 1024) -> hoist it out of the loops.
    const int which = colBase >> 10;              // 0=q 1=k 2=v
    __bf16* __restrict__ dst = (which == 0) ? qws : (which == 1) ? kws : vws;
    const float scl = (which == 0) ? 0.125f : 1.0f;   // 1/sqrt(64) folded into q

    #pragma unroll
    for (int mt = 0; mt < 4; mt++) {
        #pragma unroll
        for (int nt = 0; nt < 2; nt++) {
            const int col = colBase + wn * 32 + nt * 16 + nl;
            const int c   = col & (D_ - 1);
            const int h   = c >> 6, d = c & 63;
            float bias = 0.0f;
            if (which == 0)      bias = qb[c];
            else if (which == 2) bias = vb[c];
            #pragma unroll
            for (int r = 0; r < 8; r++) {
                const int row = rowBase + wm * 64 + mt * 16 + r + 8 * g;
                const int b   = row >> 11, s = row & (S_ - 1);
                const size_t idx = (((size_t)(b * H_ + h) * S_) + s) * HD_ + d;
                dst[idx] = f2bf((acc[mt][nt][r] + bias) * scl);
            }
        }
    }
}

// ---------------------------------------------------------------------------
// Kernel 2: flash attention, one WG per (b,h,128 q rows), 8 waves x 16 q rows
// ---------------------------------------------------------------------------
__global__ __launch_bounds__(256)
void attn_wmma(const __bf16* __restrict__ qws, const __bf16* __restrict__ kws,
               const __bf16* __restrict__ vws, float* __restrict__ out)
{
    __shared__ __bf16 lK[32][72];        // [key][d]   (filled by async copy)
    __shared__ __bf16 lV[64][40];        // [d][key]   (transposed)
    __shared__ __bf16 lP[8][16][32];     // per-wave probs scratch

    const int t = threadIdx.x;
    const int w = t >> 5, lane = t & 31, g = lane >> 4, nl = lane & 15;
    const int qt = blockIdx.x, h = blockIdx.y, b = blockIdx.z;
    const size_t bh = (size_t)(b * H_ + h) * S_ * HD_;
    const __bf16* Qp = qws + bh;
    const __bf16* Kp = kws + bh;
    const __bf16* Vp = vws + bh;

    const int q0 = qt * 128 + w * 16;

    // Q fragments for 16x64 tile (2 K-steps of 32), loaded once
    v16bf qf[2];
    #pragma unroll
    for (int kb = 0; kb < 2; kb++)
        qf[kb] = ldfrag(Qp + (size_t)(q0 + nl) * HD_ + kb * 32 + g * 8);

    const v8f vzero = {0.f,0.f,0.f,0.f,0.f,0.f,0.f,0.f};
    v8f cacc[4];
    #pragma unroll
    for (int nt = 0; nt < 4; nt++) cacc[nt] = vzero;
    float mrow[8], lrow[8];
    #pragma unroll
    for (int r = 0; r < 8; r++) { mrow[r] = -3.0e38f; lrow[r] = 0.f; }

    const int krow = t >> 3, koff = (t & 7) * 8;
    // per-lane LDS byte offset for the async K-tile fill (low 32 bits of the
    // generic LDS pointer are the LDS offset per the flat-aperture mapping)
    const unsigned ldsK = (unsigned)(size_t)&lK[krow][koff];

    for (int kt = 0; kt < S_ / 32; kt++) {
        __syncthreads();
        const size_t gro = (size_t)(kt * 32 + krow) * HD_ + koff;

        // K tile: 16B/lane global -> LDS via the CDNA5 async path (ASYNCcnt)
        const unsigned long long gk = (unsigned long long)(size_t)(Kp + gro);
        asm volatile("global_load_async_to_lds_b128 %0, %1, off"
                     :: "v"(ldsK), "v"(gk) : "memory");

        // V tile needs a transpose scatter -> register path
        const v8bf vv = *(const v8bf*)(Vp + gro);
        #pragma unroll
        for (int j = 0; j < 8; j++) lV[koff + j][krow] = vv[j];

        if (kt + 1 < S_ / 32) {   // global_prefetch_b8 for next tile
            __builtin_prefetch(Kp + gro + 32 * HD_, 0, 0);
            __builtin_prefetch(Vp + gro + 32 * HD_, 0, 0);
        }

        asm volatile("s_wait_asynccnt 0" ::: "memory");
        __syncthreads();

        // scores: 16q x 32k = 2 sub-tiles, K-dim 64 = 2 chained WMMAs each
        v8f s0 = vzero, s1 = vzero;
        #pragma unroll
        for (int kb = 0; kb < 2; kb++) {
            s0 = __builtin_amdgcn_wmma_f32_16x16x32_bf16(
                false, qf[kb], false, ldfrag(&lK[nl][kb * 32 + g * 8]),
                (short)0, s0, false, false);
            s1 = __builtin_amdgcn_wmma_f32_16x16x32_bf16(
                false, qf[kb], false, ldfrag(&lK[16 + nl][kb * 32 + g * 8]),
                (short)0, s1, false, false);
        }

        // online softmax (fp32): row M = r + 8*g lives across 16 lanes of a half
        #pragma unroll
        for (int r = 0; r < 8; r++) {
            float tmax = fmaxf(s0[r], s1[r]);
            #pragma unroll
            for (int d = 1; d < 16; d <<= 1) tmax = fmaxf(tmax, __shfl_xor(tmax, d, 32));
            const float mnew  = fmaxf(mrow[r], tmax);
            const float alpha = __expf(mrow[r] - mnew);
            mrow[r] = mnew;
            const float p0 = __expf(s0[r] - mnew);
            const float p1 = __expf(s1[r] - mnew);
            float rs = p0 + p1;
            #pragma unroll
            for (int d = 1; d < 16; d <<= 1) rs += __shfl_xor(rs, d, 32);
            lrow[r] = lrow[r] * alpha + rs;
            #pragma unroll
            for (int nt = 0; nt < 4; nt++) cacc[nt][r] *= alpha;
            lP[w][r + 8 * g][nl]      = f2bf(p0);
            lP[w][r + 8 * g][16 + nl] = f2bf(p1);
        }

        asm volatile("s_wait_dscnt 0" ::: "memory");  // wave-local LDS RAW

        // ctx += P(16x32) x V(32x64): 4 WMMAs
        const v16bf pf = ldfrag(&lP[w][nl][g * 8]);
        #pragma unroll
        for (int nt = 0; nt < 4; nt++) {
            const v16bf vf = ldfrag(&lV[nt * 16 + nl][g * 8]);
            cacc[nt] = __builtin_amdgcn_wmma_f32_16x16x32_bf16(
                false, pf, false, vf, (short)0, cacc[nt], false, false);
        }
    }

    // normalize and store fp32 to (B,S,D)
    #pragma unroll
    for (int nt = 0; nt < 4; nt++) {
        #pragma unroll
        for (int r = 0; r < 8; r++) {
            const int s = q0 + r + 8 * g;
            out[((size_t)(b * S_ + s)) * D_ + h * HD_ + nt * 16 + nl] =
                cacc[nt][r] / lrow[r];
        }
    }
}

extern "C" void kernel_launch(void* const* d_in, const int* in_sizes, int n_in,
                              void* d_out, int out_size, void* d_ws, size_t ws_size,
                              hipStream_t stream) {
    const float* X  = (const float*)d_in[0];
    const float* W  = (const float*)d_in[1];
    const float* qb = (const float*)d_in[2];
    const float* vb = (const float*)d_in[3];
    float* out = (float*)d_out;

    const size_t per = (size_t)B_ * H_ * S_ * HD_;   // 4M elems -> 8MB bf16 each
    __bf16* qws = (__bf16*)d_ws;
    __bf16* kws = qws + per;
    __bf16* vws = kws + per;

    dim3 g1((3 * D_) / 128, (B_ * S_) / 128);        // 24 x 32 blocks
    qkv_gemm_wmma<<<g1, 256, 0, stream>>>(X, W, qb, vb, qws, kws, vws);

    dim3 g2(S_ / 128, H_, B_);                       // 16 x 16 x 2 blocks
    attn_wmma<<<g2, 256, 0, stream>>>(qws, kws, vws, out);
}
